// GATNet_45844480917958
// MI455X (gfx1250) — matrix operand order repaired and checked by reading, not compiled, not run on hardware
//
#include <hip/hip_runtime.h>
#include <stdint.h>

// ---------------------------------------------------------------------------
// GATNet forward for MI455X (gfx1250): bf16 WMMA GEMMs + 3-pass edge softmax.
// ---------------------------------------------------------------------------

typedef __attribute__((ext_vector_type(16))) __bf16 v16bf;
typedef __attribute__((ext_vector_type(8)))  __bf16 v8bf;
typedef __attribute__((ext_vector_type(8)))  float  v8f;

#define LRELU_SLOPE 0.2f

// Monotonic float<->uint mapping so segment-max can use integer atomicMax.
__device__ __forceinline__ unsigned fkey(float f) {
    int i = __float_as_int(f);
    return (unsigned)(i ^ ((i >> 31) | 0x80000000));
}
__device__ __forceinline__ float funkey(unsigned k) {
    int m = (((int)(k ^ 0x80000000u)) >> 31) | 0x80000000;
    return __int_as_float((int)k ^ m);
}

// ---------------- f32 -> bf16 conversion ----------------
__global__ void k_f32_to_bf16(const float* __restrict__ src,
                              __bf16* __restrict__ dst, int n) {
    int i = blockIdx.x * blockDim.x + threadIdx.x;
    if (i < n) dst[i] = (__bf16)src[i];
}

// ---------------- WMMA GEMM: C[M,N] = A[M,K] * B[K,N] ----------------
// A,B bf16 row-major; C f32 row-major. One wave -> 16x64 output strip.
// blockDim.x = 256 (8 waves). Requires M%16==0, N%64==0, K%32==0.
__global__ void k_wmma_gemm(const __bf16* __restrict__ A,
                            const __bf16* __restrict__ B,
                            float* __restrict__ C, int M, int N, int K) {
    int wave = blockIdx.x * (blockDim.x >> 5) + (threadIdx.x >> 5);
    int lane = threadIdx.x & 31;
    int tilesN = N >> 6;
    int tm = wave / tilesN;
    int tn = wave % tilesN;
    if (tm >= (M >> 4)) return;          // uniform per wave -> EXEC stays all-1s
    int mBase = tm << 4;
    int nBase = tn << 6;
    int half  = lane >> 4;
    int mrow  = mBase + (lane & 15);

    v8f acc0 = {}; v8f acc1 = {}; v8f acc2 = {}; v8f acc3 = {};
    for (int kk = 0; kk < K; kk += 32) {
        // A fragment: lanes 0-15 hold K = kk+{0..7,16..23}; lanes 16-31 +8.
        const __bf16* arow = A + (size_t)mrow * K + kk + half * 8;
        v8bf alo = *(const v8bf*)(arow);
        v8bf ahi = *(const v8bf*)(arow + 16);
        v16bf a = __builtin_shufflevector(alo, ahi,
                                          0,1,2,3,4,5,6,7,8,9,10,11,12,13,14,15);
        // B fragment: lane holds column N = lane%16, K = kk+{0..7,16..23} (+8 for lanes 16-31).
        const __bf16* bcol = B + (size_t)(kk + half * 8) * N + nBase + (lane & 15);
        v16bf b0, b1, b2, b3;
#pragma unroll
        for (int s = 0; s < 16; ++s) {
            const __bf16* bp = bcol + (size_t)((s < 8) ? s : s + 8) * N;
            b0[s] = bp[0];
            b1[s] = bp[16];
            b2[s] = bp[32];
            b3[s] = bp[48];
        }
        acc0 = __builtin_amdgcn_wmma_f32_16x16x32_bf16(false, a, false, b0, (short)0, acc0, false, false);
        acc1 = __builtin_amdgcn_wmma_f32_16x16x32_bf16(false, a, false, b1, (short)0, acc1, false, false);
        acc2 = __builtin_amdgcn_wmma_f32_16x16x32_bf16(false, a, false, b2, (short)0, acc2, false, false);
        acc3 = __builtin_amdgcn_wmma_f32_16x16x32_bf16(false, a, false, b3, (short)0, acc3, false, false);
    }
    int col = lane & 15;
#pragma unroll
    for (int r = 0; r < 8; ++r) {
        int row = mBase + r + half * 8;
        float* crow = C + (size_t)row * N + col;
        crow[nBase +  0] = acc0[r];
        crow[nBase + 16] = acc1[r];
        crow[nBase + 32] = acc2[r];
        crow[nBase + 48] = acc3[r];
    }
}

// ---------------- per-node attention dot products ----------------
// al_s[n*H+h] = sum_c h[n,h,c]*a_src[h,c]; same for al_d. One wave per (n,h).
__global__ void k_node_alpha(const float* __restrict__ h,
                             const float* __restrict__ a_src,
                             const float* __restrict__ a_dst,
                             float* __restrict__ al_s, float* __restrict__ al_d,
                             int n, int H, int D) {
    int wave = blockIdx.x * (blockDim.x >> 5) + (threadIdx.x >> 5);
    int lane = threadIdx.x & 31;
    if (wave >= n * H) return;
    int node = wave / H, head = wave % H;
    const float* hp = h + (size_t)node * H * D + (size_t)head * D;
    const float* as = a_src + head * D;
    const float* ad = a_dst + head * D;
    float ss = 0.f, sd = 0.f;
    for (int c = lane; c < D; c += 32) {
        float v = hp[c];
        ss += v * as[c];
        sd += v * ad[c];
    }
#pragma unroll
    for (int off = 16; off > 0; off >>= 1) {
        ss += __shfl_xor(ss, off, 32);
        sd += __shfl_xor(sd, off, 32);
    }
    if (lane == 0) { al_s[wave] = ss; al_d[wave] = sd; }
}

// ---------------- edge pass 1: leaky-relu score + segment max ----------------
__global__ void k_edge_max(const int* __restrict__ src, const int* __restrict__ dst,
                           int E, int Etot,
                           const float* __restrict__ al_s, const float* __restrict__ al_d,
                           float* __restrict__ score, unsigned* __restrict__ mkey, int H) {
    int idx = blockIdx.x * blockDim.x + threadIdx.x;
    if (idx >= Etot * H) return;
    int e = idx / H, hh = idx % H;
    int s = (e < E) ? src[e] : (e - E);   // edges then self-loops
    int d = (e < E) ? dst[e] : (e - E);
    float sc = al_s[s * H + hh] + al_d[d * H + hh];
    sc = (sc > 0.f) ? sc : LRELU_SLOPE * sc;
    score[idx] = sc;
    atomicMax(&mkey[d * H + hh], fkey(sc));
}

// ---------------- edge pass 2: exp(score - max) + segment sum ----------------
__global__ void k_edge_exp(const int* __restrict__ dst, int E, int Etot,
                           float* __restrict__ score /* in: score, out: exp */,
                           const unsigned* __restrict__ mkey,
                           float* __restrict__ denom, int H) {
    int idx = blockIdx.x * blockDim.x + threadIdx.x;
    if (idx >= Etot * H) return;
    int e = idx / H, hh = idx % H;
    int d = (e < E) ? dst[e] : (e - E);
    float ex = __expf(score[idx] - funkey(mkey[d * H + hh]));
    score[idx] = ex;
    atomicAdd(&denom[d * H + hh], ex);
}

// ---------------- edge pass 3: alpha-weighted scatter-add ----------------
// One block per edge; threads stride H*D channels.
__global__ void k_edge_scatter(const int* __restrict__ src, const int* __restrict__ dst,
                               int E,
                               const float* __restrict__ h, const float* __restrict__ ex,
                               const float* __restrict__ denom, float* __restrict__ agg,
                               int H, int D) {
    int e = blockIdx.x;
    int s = (e < E) ? src[e] : (e - E);
    int d = (e < E) ? dst[e] : (e - E);
    int HD = H * D;
    const float* hs = h + (size_t)s * HD;
    float* ag = agg + (size_t)d * HD;
    for (int c = threadIdx.x; c < HD; c += blockDim.x) {
        int hh = c / D;
        float alpha = ex[(size_t)e * H + hh] / denom[d * H + hh];
        atomicAdd(&ag[c], hs[c] * alpha);
    }
}

// ---------------- epilogue 1: +bias, ELU, emit bf16 for GEMM2 ----------------
__global__ void k_epi1(const float* __restrict__ agg, const float* __restrict__ b,
                       __bf16* __restrict__ x1bf, int n, int HD) {
    int i = blockIdx.x * blockDim.x + threadIdx.x;
    if (i >= n * HD) return;
    float o = agg[i] + b[i % HD];
    float v = (o > 0.f) ? o : (__expf(o) - 1.f);
    x1bf[i] = (__bf16)v;
}

// ---------------- epilogue 2: +bias, ReLU in place ----------------
__global__ void k_epi2(float* __restrict__ agg, const float* __restrict__ b,
                       int n, int Dd) {
    int i = blockIdx.x * blockDim.x + threadIdx.x;
    if (i >= n * Dd) return;
    float o = agg[i] + b[i % Dd];
    agg[i] = (o > 0.f) ? o : 0.f;
}

// ---------------- global max pool over graphs ----------------
__global__ void k_pool(const float* __restrict__ x2, const int* __restrict__ batch,
                       unsigned* __restrict__ pkey, int n, int Dd) {
    int i = blockIdx.x * blockDim.x + threadIdx.x;
    if (i >= n * Dd) return;
    int node = i / Dd, c = i % Dd;
    atomicMax(&pkey[(size_t)batch[node] * Dd + c], fkey(x2[i]));
}

// ---------------- FC head: 128 -> 16 (ReLU) -> 1 ----------------
__global__ void k_fc(const unsigned* __restrict__ pkey,
                     const float* __restrict__ fc1_w, const float* __restrict__ fc1_b,
                     const float* __restrict__ fc2_w, const float* __restrict__ fc2_b,
                     float* __restrict__ out, int G, int Dd) {
    int g = blockIdx.x * blockDim.x + threadIdx.x;
    if (g >= G) return;
    float acc[16];
#pragma unroll
    for (int j = 0; j < 16; ++j) acc[j] = fc1_b[j];
    for (int c = 0; c < Dd; ++c) {
        float p = fmaxf(funkey(pkey[(size_t)g * Dd + c]), 0.0f); // clamp handles empty segs
#pragma unroll
        for (int j = 0; j < 16; ++j) acc[j] += p * fc1_w[c * 16 + j];
    }
    float o = fc2_b[0];
#pragma unroll
    for (int j = 0; j < 16; ++j) {
        float hj = (acc[j] > 0.f) ? acc[j] : 0.f;
        o += hj * fc2_w[j];
    }
    out[g] = o;
}

// ---------------------------------------------------------------------------
extern "C" void kernel_launch(void* const* d_in, const int* in_sizes, int n_in,
                              void* d_out, int out_size, void* d_ws, size_t ws_size,
                              hipStream_t stream) {
    const float* x      = (const float*)d_in[0];
    const int*   eidx   = (const int*)d_in[1];
    const int*   batch  = (const int*)d_in[2];
    const float* W1     = (const float*)d_in[3];
    const float* a_src1 = (const float*)d_in[4];
    const float* a_dst1 = (const float*)d_in[5];
    const float* b1     = (const float*)d_in[6];
    const float* W2     = (const float*)d_in[7];
    const float* a_src2 = (const float*)d_in[8];
    const float* a_dst2 = (const float*)d_in[9];
    const float* b2     = (const float*)d_in[10];
    const float* fc1_w  = (const float*)d_in[11];
    const float* fc1_b  = (const float*)d_in[12];
    const float* fc2_w  = (const float*)d_in[13];
    const float* fc2_b  = (const float*)d_in[14];
    float* out = (float*)d_out;
    (void)n_in; (void)ws_size;

    const int D = 128, H = 4;
    const int N    = in_sizes[0] / D;       // 20000
    const int E    = in_sizes[1] / 2;       // 320000
    const int Etot = E + N;                 // edges + self loops
    const int G    = out_size;              // 256
    const int HD   = H * D;                 // 512
    const int* src = eidx;
    const int* dst = eidx + E;

    // ---- carve workspace (buffers reused across layers) ----
    char* p = (char*)d_ws;
    auto take = [&](size_t bytes) -> char* {
        char* r = p;
        p += (bytes + 255) & ~(size_t)255;
        return r;
    };
    __bf16*   xbf   = (__bf16*)  take((size_t)N * HD * 2); // x bf16, later x1 bf16
    __bf16*   w1bf  = (__bf16*)  take((size_t)D * HD * 2);
    __bf16*   w2bf  = (__bf16*)  take((size_t)HD * D * 2);
    float*    h1    = (float*)   take((size_t)N * HD * 4); // later h2
    float*    agg   = (float*)   take((size_t)N * HD * 4); // later agg2 / x2
    float*    als   = (float*)   take((size_t)N * H * 4);
    float*    ald   = (float*)   take((size_t)N * H * 4);
    unsigned* mk    = (unsigned*)take((size_t)N * H * 4);
    float*    den   = (float*)   take((size_t)N * H * 4);
    float*    score = (float*)   take((size_t)Etot * H * 4);
    unsigned* pk    = (unsigned*)take((size_t)G * D * 4);

    // ---- precision conversion for WMMA ----
    k_f32_to_bf16<<<(N * D + 255) / 256, 256, 0, stream>>>(x, xbf, N * D);
    k_f32_to_bf16<<<(D * HD + 255) / 256, 256, 0, stream>>>(W1, w1bf, D * HD);
    k_f32_to_bf16<<<(HD * D + 255) / 256, 256, 0, stream>>>(W2, w2bf, HD * D);

    // ======================= layer 1 =======================
    {
        int tiles = (N / 16) * (HD / 64);
        k_wmma_gemm<<<(tiles + 7) / 8, 256, 0, stream>>>(xbf, w1bf, h1, N, HD, D);
    }
    k_node_alpha<<<(N * H + 7) / 8, 256, 0, stream>>>(h1, a_src1, a_dst1, als, ald, N, H, D);
    hipMemsetAsync(mk, 0, (size_t)N * H * 4, stream);
    hipMemsetAsync(den, 0, (size_t)N * H * 4, stream);
    hipMemsetAsync(agg, 0, (size_t)N * HD * 4, stream);
    k_edge_max<<<((size_t)Etot * H + 255) / 256, 256, 0, stream>>>(src, dst, E, Etot, als, ald, score, mk, H);
    k_edge_exp<<<((size_t)Etot * H + 255) / 256, 256, 0, stream>>>(dst, E, Etot, score, mk, den, H);
    k_edge_scatter<<<Etot, 256, 0, stream>>>(src, dst, E, h1, score, den, agg, H, D);
    k_epi1<<<((size_t)N * HD + 255) / 256, 256, 0, stream>>>(agg, b1, xbf, N, HD);

    // ======================= layer 2 (heads = 1) =======================
    {
        int tiles = (N / 16) * (D / 64);
        k_wmma_gemm<<<(tiles + 7) / 8, 256, 0, stream>>>(xbf, w2bf, h1, N, D, HD);
    }
    k_node_alpha<<<(N + 7) / 8, 256, 0, stream>>>(h1, a_src2, a_dst2, als, ald, N, 1, D);
    hipMemsetAsync(mk, 0, (size_t)N * 4, stream);
    hipMemsetAsync(den, 0, (size_t)N * 4, stream);
    hipMemsetAsync(agg, 0, (size_t)N * D * 4, stream);
    k_edge_max<<<((size_t)Etot + 255) / 256, 256, 0, stream>>>(src, dst, E, Etot, als, ald, score, mk, 1);
    k_edge_exp<<<((size_t)Etot + 255) / 256, 256, 0, stream>>>(dst, E, Etot, score, mk, den, 1);
    k_edge_scatter<<<Etot, 128, 0, stream>>>(src, dst, E, h1, score, den, agg, 1, D);
    k_epi2<<<((size_t)N * D + 255) / 256, 256, 0, stream>>>(agg, b2, N, D);

    // ======================= pool + FC head =======================
    hipMemsetAsync(pk, 0, (size_t)G * D * 4, stream);
    k_pool<<<((size_t)N * D + 255) / 256, 256, 0, stream>>>(agg, batch, pk, N, D);
    k_fc<<<(G + 63) / 64, 64, 0, stream>>>(pk, fc1_w, fc1_b, fc2_w, fc2_b, out, G, D);
}
